// EnergyTransformer_59476707115408
// MI455X (gfx1250) — matrix operand order repaired and checked by reading, not compiled, not run on hardware
//
#include <hip/hip_runtime.h>
#include <cstdint>

// ---------------------------------------------------------------------------
// Types
// ---------------------------------------------------------------------------
typedef __bf16 bf16_t;
typedef __attribute__((ext_vector_type(16))) __bf16 bf16x16;
typedef __attribute__((ext_vector_type(8)))  __bf16 bf16x8;
typedef __attribute__((ext_vector_type(8)))  float  f32x8;

// ---------------------------------------------------------------------------
// Problem constants
// ---------------------------------------------------------------------------
#define EMBED   768
#define NTOK    197      // patches + cls
#define NPATCH  196
#define BATCH   16
#define NHEADS  12
#define HDIM    64
#define HOPH    3072
#define TSTEPS  6
#define BETA_S  0.125f   // 1/sqrt(64)
#define EPS_LN  1e-5f

#define MTOK (BATCH*NTOK)    // 3152
#define MPAT (BATCH*NPATCH)  // 3136

// attention kernel LDS geometry
#define TPAD   224   // token dim padded to 7*32
#define TSTR   232   // score row stride (bf16 elems) -> conflict-free b128 loads
#define QKSTR  72    // Q/K row stride (bf16 elems)
#define ATT_LDS_BYTES ((2*TPAD*QKSTR + TPAD*TSTR)*2)   // 168,448 B (<320KB/WG)

// ---------------------------------------------------------------------------
// CDNA5 async global->LDS (ASYNCcnt) support, guarded so a toolchain without
// the builtin falls back to the proven synchronous path.
// Builtin signature (from hipcc diagnostic): first param is a pointer to
// __attribute__((vector_size(16))) int in the GLOBAL address space.
// ---------------------------------------------------------------------------
#if defined(__gfx1250__) && __has_builtin(__builtin_amdgcn_global_load_async_to_lds_b128)
#define HAVE_ASYNC 1
#endif

#ifdef HAVE_ASYNC
typedef int v4i_t __attribute__((vector_size(16)));
typedef __attribute__((address_space(1))) v4i_t* gv4i_p;   // global src
typedef __attribute__((address_space(3))) v4i_t* lv4i_p;   // LDS dst

static __device__ __forceinline__ void async_b128(const void* g, void* l) {
  __builtin_amdgcn_global_load_async_to_lds_b128((gv4i_p)g, (lv4i_p)l,
                                                 /*offset=*/0, /*cpol=*/0);
}
static __device__ __forceinline__ void wait_async() {
#if __has_builtin(__builtin_amdgcn_s_wait_asynccnt)
  __builtin_amdgcn_s_wait_asynccnt(0);
#else
  asm volatile("s_wait_asynccnt 0" ::: "memory");
#endif
}
#endif

// ---------------------------------------------------------------------------
// WMMA helpers (CDNA5 gfx1250, wave32)
// ---------------------------------------------------------------------------
static __device__ __forceinline__ f32x8 fzero() {
  f32x8 z = {0.f,0.f,0.f,0.f,0.f,0.f,0.f,0.f};
  return z;
}

static __device__ __forceinline__ f32x8 wmma_bf16(bf16x16 a, bf16x16 b, f32x8 c) {
  // (neg_a, A, neg_b, B, c_mod, C, reuse_a, reuse_b)
  return __builtin_amdgcn_wmma_f32_16x16x32_bf16(false, a, false, b, (short)0, c,
                                                 false, false);
}

// A fragment (16x32 bf16, row-major in LDS). rowp points to this lane's row.
// koff = 32-step offset + (lane/16)*16.  Per-lane k = koff..koff+15.
static __device__ __forceinline__ bf16x16 frag_a(const bf16_t* rowp, int koff) {
  bf16x8 lo = *(const bf16x8*)(rowp + koff);
  bf16x8 hi = *(const bf16x8*)(rowp + koff + 8);
  bf16x16 v;
#pragma unroll
  for (int j = 0; j < 8; ++j) { v[j] = lo[j]; v[j+8] = hi[j]; }
  return v;
}

// B fragment (32x16 bf16) from Bt storage [n_out][k] row-major.
// rowp points to this lane's output-column row; per-lane k = koff..koff+15
// with koff = 32-step offset + (lane/16)*16 (contiguous per lane-half).
static __device__ __forceinline__ bf16x16 frag_b(const bf16_t* rowp, int koff) {
  bf16x8 lo = *(const bf16x8*)(rowp + koff);
  bf16x8 hi = *(const bf16x8*)(rowp + koff + 8);
  bf16x16 v;
#pragma unroll
  for (int j = 0; j < 8; ++j) { v[j] = lo[j]; v[j+8] = hi[j]; }
  return v;
}

// A fragment gathered from a TRANSPOSED matrix: A[row][k] = base[k*stride+col]
static __device__ __forceinline__ bf16x16 frag_a_t(const bf16_t* base, int col,
                                                   int stride, int kbase) {
  bf16x16 v;
#pragma unroll
  for (int j = 0; j < 16; ++j) {
    int k = kbase + j;
    v[j] = base[(size_t)k*stride + col];
  }
  return v;
}

// B fragment gathered from row-major [k_inner][col] storage (strided)
static __device__ __forceinline__ bf16x16 frag_b_t(const bf16_t* base, int col,
                                                   int stride, int kbase) {
  bf16x16 v;
#pragma unroll
  for (int j = 0; j < 16; ++j) {
    v[j] = base[(size_t)(kbase + j)*stride + col];
  }
  return v;
}

// ---------------------------------------------------------------------------
// Generic bf16 WMMA GEMM:  C[M,N] = A[M,K] * Bt[N,K]^T  (+bias, relu, modes)
// block tile 128x64x32, 256 threads = 8 waves, 2x2 16x16 frags per wave.
// With async support: 2-stage LDS pipeline, tile t+1 streams via ASYNCcnt
// while tile t is consumed by the WMMAs (one barrier per K-step).
// mode: 0 normal (Cf and/or Cb), 1 accumulate into h (ldc=768),
//       2 embed row-remap into h, 3 final out (skip CLS)
// ---------------------------------------------------------------------------
__launch_bounds__(256)
__global__ void k_gemm(const bf16_t* __restrict__ A, const bf16_t* __restrict__ Bt,
                       int M, int N, int K,
                       float* __restrict__ Cf, bf16_t* __restrict__ Cb,
                       const float* __restrict__ bias, int mode, int relu)
{
  const int tid  = threadIdx.x;
  const int wave = tid >> 5, lane = tid & 31;
  const int half = lane >> 4, lr = lane & 15;
  const int wm   = wave >> 1, wn = wave & 1;
  const int m0   = blockIdx.x * 128;
  const int n0   = blockIdx.y * 64;

  const int arow = tid >> 1;           // 0..127
  const int acol = (tid & 1) * 16;     // 0 or 16
  const int brow = tid >> 2;           // 0..63
  const int bcol = (tid & 3) * 8;      // 0,8,16,24

  f32x8 acc[2][2];
  acc[0][0] = fzero(); acc[0][1] = fzero();
  acc[1][0] = fzero(); acc[1][1] = fzero();

#ifdef HAVE_ASYNC
  // ---------------- double-buffered async pipeline ----------------
  __shared__ __align__(16) bf16_t As[2][128*40];
  __shared__ __align__(16) bf16_t Bs[2][64*40];

  const int gr = m0 + arow;
  auto load_tile = [&](int k0, int buf) {
    if (gr < M) {
      const bf16_t* gA = A + (size_t)gr*K + k0 + acol;
      async_b128(gA,     &As[buf][arow*40 + acol]);
      async_b128(gA + 8, &As[buf][arow*40 + acol + 8]);
    } else {
      const uint4 z = {0,0,0,0};
      *(uint4*)&As[buf][arow*40 + acol]     = z;
      *(uint4*)&As[buf][arow*40 + acol + 8] = z;
    }
    async_b128(Bt + (size_t)(n0 + brow)*K + k0 + bcol, &Bs[buf][brow*40 + bcol]);
  };

  load_tile(0, 0);
  wait_async();
  __syncthreads();

  int buf = 0;
  for (int k0 = 0; k0 < K; k0 += 32) {
    if (k0 + 32 < K) load_tile(k0 + 32, buf ^ 1);   // stream next tile

    bf16x16 af0 = frag_a(&As[buf][(wm*32 +  0 + lr)*40], half*16);
    bf16x16 af1 = frag_a(&As[buf][(wm*32 + 16 + lr)*40], half*16);
    bf16x16 bf0 = frag_b(&Bs[buf][(wn*32 +  0 + lr)*40], half*16);
    bf16x16 bf1 = frag_b(&Bs[buf][(wn*32 + 16 + lr)*40], half*16);
    acc[0][0] = wmma_bf16(af0, bf0, acc[0][0]);
    acc[0][1] = wmma_bf16(af0, bf1, acc[0][1]);
    acc[1][0] = wmma_bf16(af1, bf0, acc[1][0]);
    acc[1][1] = wmma_bf16(af1, bf1, acc[1][1]);

    wait_async();          // own async stores to LDS done
    __syncthreads();       // everyone's done -> next buffer valid
    buf ^= 1;
  }
#else
  // ---------------- synchronous fallback ----------------
  __shared__ __align__(16) bf16_t As[128*40];
  __shared__ __align__(16) bf16_t Bs[64*40];

  for (int k0 = 0; k0 < K; k0 += 32) {
    uint4 a0 = {0,0,0,0}, a1 = {0,0,0,0};
    const int gr = m0 + arow;
    if (gr < M) {
      const uint4* src = (const uint4*)(A + (size_t)gr*K + k0 + acol);
      a0 = src[0]; a1 = src[1];
    }
    *(uint4*)&As[arow*40 + acol]     = a0;
    *(uint4*)&As[arow*40 + acol + 8] = a1;

    const uint4* bsrc = (const uint4*)(Bt + (size_t)(n0 + brow)*K + k0 + bcol);
    *(uint4*)&Bs[brow*40 + bcol] = *bsrc;

    if (k0 + 32 < K) {
      __builtin_prefetch(Bt + (size_t)(n0 + brow)*K + k0 + 32 + bcol, 0, 3);
    }
    __syncthreads();

    bf16x16 af0 = frag_a(&As[(wm*32 +  0 + lr)*40], half*16);
    bf16x16 af1 = frag_a(&As[(wm*32 + 16 + lr)*40], half*16);
    bf16x16 bf0 = frag_b(&Bs[(wn*32 +  0 + lr)*40], half*16);
    bf16x16 bf1 = frag_b(&Bs[(wn*32 + 16 + lr)*40], half*16);
    acc[0][0] = wmma_bf16(af0, bf0, acc[0][0]);
    acc[0][1] = wmma_bf16(af0, bf1, acc[0][1]);
    acc[1][0] = wmma_bf16(af1, bf0, acc[1][0]);
    acc[1][1] = wmma_bf16(af1, bf1, acc[1][1]);
    __syncthreads();
  }
#endif

  // ---- store ----
#pragma unroll
  for (int a = 0; a < 2; ++a) {
#pragma unroll
    for (int bb = 0; bb < 2; ++bb) {
      const int colg = n0 + wn*32 + bb*16 + lr;
#pragma unroll
      for (int i = 0; i < 8; ++i) {
        const int rowg = m0 + wm*32 + a*16 + half*8 + i;
        if (rowg >= M) continue;
        float v = acc[a][bb][i];
        if (bias) v += bias[colg];
        if (relu) v = fmaxf(v, 0.f);
        if (mode == 0) {
          if (Cf) Cf[(size_t)rowg*N + colg] = v;
          if (Cb) Cb[(size_t)rowg*N + colg] = (bf16_t)v;
        } else if (mode == 1) {
          Cf[(size_t)rowg*EMBED + colg] += v;
        } else if (mode == 2) {               // tok rows -> h rows (t>=1)
          const int bi = rowg / NPATCH, t = rowg % NPATCH;
          Cf[((size_t)bi*NTOK + 1 + t)*EMBED + colg] = v;
        } else {                              // final: skip CLS row
          const int bi = rowg / NTOK, t = rowg % NTOK;
          if (t > 0) Cf[((size_t)bi*NPATCH + (t-1))*EMBED + colg] = v;
        }
      }
    }
  }
}

// ---------------------------------------------------------------------------
// Fused attention-energy gradient, one block per (batch, head).
// QK bf16 [3152][1536]  (cols 0:768 = Q heads, 768:1536 = K heads)
// G  bf16 [3152][1536]  (cols 0:768 = P K, 768:1536 = P^T Q)
// ---------------------------------------------------------------------------
__launch_bounds__(256)
__global__ void k_attention(const bf16_t* __restrict__ QK, bf16_t* __restrict__ G)
{
  extern __shared__ bf16_t smem[];
  bf16_t* Qs = smem;                       // [224][72]
  bf16_t* Ks = smem + TPAD*QKSTR;          // [224][72]
  bf16_t* Ps = smem + 2*TPAD*QKSTR;        // [224][232]  scores -> probs

  const int tid  = threadIdx.x;
  const int wave = tid >> 5, lane = tid & 31;
  const int half = lane >> 4, lr = lane & 15;
  const int b    = blockIdx.x / NHEADS;
  const int hd   = blockIdx.x % NHEADS;

  // Phase 0: load Q,K head slices (zero-pad rows >= 197)
  for (int e = tid; e < TPAD*8; e += 256) {
    const int t  = e >> 3;
    const int kc = (e & 7) * 8;
#ifdef HAVE_ASYNC
    if (t < NTOK) {
      const size_t row = (size_t)(b*NTOK + t)*1536;
      async_b128(QK + row + hd*HDIM + kc,       &Qs[t*QKSTR + kc]);
      async_b128(QK + row + 768 + hd*HDIM + kc, &Ks[t*QKSTR + kc]);
    } else {
      const uint4 z = {0,0,0,0};
      *(uint4*)&Qs[t*QKSTR + kc] = z;
      *(uint4*)&Ks[t*QKSTR + kc] = z;
    }
#else
    uint4 vq = {0,0,0,0}, vk = {0,0,0,0};
    if (t < NTOK) {
      const size_t row = (size_t)(b*NTOK + t)*1536;
      vq = *(const uint4*)(QK + row + hd*HDIM + kc);
      vk = *(const uint4*)(QK + row + 768 + hd*HDIM + kc);
    }
    *(uint4*)&Qs[t*QKSTR + kc] = vq;
    *(uint4*)&Ks[t*QKSTR + kc] = vk;
#endif
  }
#ifdef HAVE_ASYNC
  wait_async();
#endif
  __syncthreads();

  // Phase 1: S[m][n] = sum_k Q[m,k] K[n,k]  (14x14 tiles of 16x16)
  for (int tile = wave; tile < 14*14; tile += 8) {
    const int mt = tile / 14, nt = tile % 14;
    f32x8 acc = fzero();
#pragma unroll
    for (int kk = 0; kk < 64; kk += 32) {
      bf16x16 a  = frag_a(&Qs[(mt*16 + lr)*QKSTR], kk + half*16);
      bf16x16 bb = frag_b(&Ks[(nt*16 + lr)*QKSTR], kk + half*16);
      acc = wmma_bf16(a, bb, acc);
    }
#pragma unroll
    for (int i = 0; i < 8; ++i) {
      const int m = mt*16 + half*8 + i;
      const int n = nt*16 + lr;
      Ps[m*TSTR + n] = (bf16_t)acc[i];
    }
  }
  __syncthreads();

  // Phase 2: softmax over keys n for each query row m (wave per row)
  for (int m = wave; m < TPAD; m += 8) {
    float e[8];
    float mx = -1e30f;
#pragma unroll
    for (int i = 0; i < 8; ++i) {
      const int n = lane + i*32;
      const float v = (n < NTOK) ? (float)Ps[m*TSTR + n] : -1e30f;
      e[i] = v;
      mx = fmaxf(mx, v);
    }
    for (int off = 16; off; off >>= 1) mx = fmaxf(mx, __shfl_xor(mx, off, 32));
    float s = 0.f;
#pragma unroll
    for (int i = 0; i < 8; ++i) {
      const int n = lane + i*32;
      const float ev = (n < NTOK) ? __expf(BETA_S*(e[i] - mx)) : 0.f;
      e[i] = ev;
      s += ev;
    }
    for (int off = 16; off; off >>= 1) s += __shfl_xor(s, off, 32);
    const float inv = 1.f / s;
    const bool rowok = (m < NTOK);
#pragma unroll
    for (int i = 0; i < 8; ++i) {
      const int n = lane + i*32;
      if (n < TSTR) {
        const float pv = (rowok && n < NTOK) ? e[i]*inv : 0.f;
        Ps[m*TSTR + n] = (bf16_t)pv;
      }
    }
  }
  __syncthreads();

  // Phase 3a: dQhat[m][k] = sum_n P[m][n] K[n][k]   (13 m-tiles x 4 k-tiles)
  for (int tile = wave; tile < 13*4; tile += 8) {
    const int mt = tile >> 2, kt = tile & 3;
    f32x8 acc = fzero();
    for (int nn = 0; nn < TPAD; nn += 32) {
      bf16x16 a  = frag_a(&Ps[(mt*16 + lr)*TSTR], nn + half*16);
      bf16x16 bb = frag_b_t(Ks, kt*16 + lr, QKSTR, nn + half*16);
      acc = wmma_bf16(a, bb, acc);
    }
#pragma unroll
    for (int i = 0; i < 8; ++i) {
      const int m = mt*16 + half*8 + i;
      if (m < NTOK)
        G[((size_t)(b*NTOK + m))*1536 + hd*HDIM + kt*16 + lr] = (bf16_t)acc[i];
    }
  }

  // Phase 3b: dKhat[n][k] = sum_m P[m][n] Q[m][k]   (A = P^T, strided gather)
  for (int tile = wave; tile < 13*4; tile += 8) {
    const int nt = tile >> 2, kt = tile & 3;
    f32x8 acc = fzero();
    for (int mm = 0; mm < TPAD; mm += 32) {
      bf16x16 a  = frag_a_t(Ps, nt*16 + lr, TSTR, mm + half*16);
      bf16x16 bb = frag_b_t(Qs, kt*16 + lr, QKSTR, mm + half*16);
      acc = wmma_bf16(a, bb, acc);
    }
#pragma unroll
    for (int i = 0; i < 8; ++i) {
      const int n = nt*16 + half*8 + i;
      if (n < NTOK)
        G[((size_t)(b*NTOK + n))*1536 + 768 + hd*HDIM + kt*16 + lr] = (bf16_t)acc[i];
    }
  }
}

// ---------------------------------------------------------------------------
// LayerNorm (wave per 768-wide row) -> bf16
// ---------------------------------------------------------------------------
__launch_bounds__(256)
__global__ void k_layernorm(const float* __restrict__ X, const float* __restrict__ gamma,
                            const float* __restrict__ beta, bf16_t* __restrict__ Y,
                            int rows)
{
  const int wave = threadIdx.x >> 5, lane = threadIdx.x & 31;
  const int r = blockIdx.x*8 + wave;
  if (r >= rows) return;
  const float* x = X + (size_t)r*EMBED;
  float v[24];
  float s = 0.f;
#pragma unroll
  for (int i = 0; i < 24; ++i) { v[i] = x[lane + i*32]; s += v[i]; }
  for (int off = 16; off; off >>= 1) s += __shfl_xor(s, off, 32);
  const float mean = s * (1.f/EMBED);
  float q = 0.f;
#pragma unroll
  for (int i = 0; i < 24; ++i) { const float d = v[i]-mean; q += d*d; }
  for (int off = 16; off; off >>= 1) q += __shfl_xor(q, off, 32);
  const float inv = rsqrtf(q*(1.f/EMBED) + EPS_LN);
#pragma unroll
  for (int i = 0; i < 24; ++i) {
    const int n = lane + i*32;
    Y[(size_t)r*EMBED + n] = (bf16_t)(gamma[n]*(v[i]-mean)*inv + beta[n]);
  }
}

// ---------------------------------------------------------------------------
// Setup / packing kernels
// ---------------------------------------------------------------------------
__global__ void k_transpose_bf16(const float* __restrict__ src, bf16_t* __restrict__ dst,
                                 int IN, int OUT)  // src[IN][OUT] -> dst[OUT][IN]
{
  const int id = blockIdx.x*blockDim.x + threadIdx.x;
  if (id >= IN*OUT) return;
  const int o = id / IN, i = id % IN;
  dst[id] = (bf16_t)src[(size_t)i*OUT + o];
}

__global__ void k_copy_bf16(const float* __restrict__ src, bf16_t* __restrict__ dst, int n)
{
  const int id = blockIdx.x*blockDim.x + threadIdx.x;
  if (id < n) dst[id] = (bf16_t)src[id];
}

// wq,wk [12][768][64] -> Wf [1536][768] (Bt for fwd) and Wb [768][1536] (Bt for bwd)
__global__ void k_pack_wqk(const float* __restrict__ wq, const float* __restrict__ wk,
                           bf16_t* __restrict__ Wf, bf16_t* __restrict__ Wb)
{
  const int id = blockIdx.x*blockDim.x + threadIdx.x;
  if (id >= 1536*EMBED) return;
  const int o = id / EMBED, d = id % EMBED;
  const int isK = (o >= 768);
  const int oh = o & 767;
  const int hh = oh / 64, kk = oh % 64;
  const float v = (isK ? wk : wq)[((size_t)hh*EMBED + d)*64 + kk];
  Wf[(size_t)o*EMBED + d] = (bf16_t)v;
  Wb[(size_t)d*1536 + o]  = (bf16_t)v;
}

// patchify x [16][3][224][224] -> bf16 [3136][768]
__global__ void k_patchify(const float* __restrict__ x, bf16_t* __restrict__ Pm)
{
  const int id = blockIdx.x*blockDim.x + threadIdx.x;
  if (id >= MPAT*EMBED) return;
  const int col = id % EMBED, r = id / EMBED;
  const int b = r / NPATCH, pp = r % NPATCH;
  const int ph = pp / 14, pw = pp % 14;
  const int c = col / 256, py = (col % 256) / 16, px = col % 16;
  const float v = x[(((size_t)b*3 + c)*224 + (ph*16 + py))*224 + (pw*16 + px)];
  Pm[id] = (bf16_t)v;
}

// set cls row, add positional embedding
__global__ void k_init_h(float* __restrict__ h, const float* __restrict__ cls,
                         const float* __restrict__ pos)
{
  const int id = blockIdx.x*blockDim.x + threadIdx.x;
  if (id >= MTOK*EMBED) return;
  const int d = id % EMBED;
  const int t = (id / EMBED) % NTOK;
  if (t == 0) h[id] = cls[d] + pos[d];
  else        h[id] += pos[(size_t)t*EMBED + d];
}

// ---------------------------------------------------------------------------
// Host launcher
// ---------------------------------------------------------------------------
static void launch_gemm(const bf16_t* A, const bf16_t* Bt, int M, int N, int K,
                        float* Cf, bf16_t* Cb, const float* bias,
                        int mode, int relu, hipStream_t s)
{
  dim3 grid((M + 127)/128, N/64);
  k_gemm<<<grid, 256, 0, s>>>(A, Bt, M, N, K, Cf, Cb, bias, mode, relu);
}

extern "C" void kernel_launch(void* const* d_in, const int* in_sizes, int n_in,
                              void* d_out, int out_size, void* d_ws, size_t ws_size,
                              hipStream_t stream)
{
  const float* x         = (const float*)d_in[0];
  const float* patch_w   = (const float*)d_in[1];
  const float* patch_b   = (const float*)d_in[2];
  const float* cls_tok   = (const float*)d_in[3];
  const float* pos_emb   = (const float*)d_in[4];
  const float* ln_gamma  = (const float*)d_in[5];
  const float* ln_beta   = (const float*)d_in[6];
  const float* wq        = (const float*)d_in[7];
  const float* wk        = (const float*)d_in[8];
  const float* xi        = (const float*)d_in[9];
  const float* out_gamma = (const float*)d_in[10];
  const float* out_beta  = (const float*)d_in[11];
  const float* out_w     = (const float*)d_in[12];
  const float* out_b     = (const float*)d_in[13];
  float* out = (float*)d_out;
  (void)in_sizes; (void)n_in; (void)out_size; (void)ws_size;

  // ---- carve workspace ----
  char* p = (char*)d_ws;
  auto carve = [&](size_t bytes) -> char* {
    char* r = p;
    p += (bytes + 255) & ~(size_t)255;
    return r;
  };
  float*  h       = (float*) carve((size_t)MTOK*EMBED*4);
  bf16_t* g       = (bf16_t*)carve((size_t)MTOK*EMBED*2);
  bf16_t* QK      = (bf16_t*)carve((size_t)MTOK*1536*2);
  bf16_t* Hhid    = (bf16_t*)carve((size_t)MTOK*HOPH*2);
  bf16_t* G       = (bf16_t*)carve((size_t)MTOK*1536*2);
  bf16_t* Pm      = (bf16_t*)carve((size_t)MPAT*EMBED*2);
  bf16_t* patchWt = (bf16_t*)carve((size_t)EMBED*EMBED*2);
  bf16_t* outWt   = (bf16_t*)carve((size_t)EMBED*EMBED*2);
  bf16_t* WqkF    = (bf16_t*)carve((size_t)1536*EMBED*2);
  bf16_t* WqkB    = (bf16_t*)carve((size_t)1536*EMBED*2);
  bf16_t* xiTf    = (bf16_t*)carve((size_t)EMBED*HOPH*2);
  bf16_t* xiB     = (bf16_t*)carve((size_t)EMBED*HOPH*2);

  // allow large dynamic LDS for the attention kernel (harmless if redundant)
  (void)hipFuncSetAttribute((const void*)k_attention,
                            hipFuncAttributeMaxDynamicSharedMemorySize,
                            ATT_LDS_BYTES);

  // ---- weight packing (bf16) ----
  const int n1 = EMBED*EMBED;
  k_transpose_bf16<<<(n1+255)/256, 256, 0, stream>>>(patch_w, patchWt, EMBED, EMBED);
  k_transpose_bf16<<<(n1+255)/256, 256, 0, stream>>>(out_w, outWt, EMBED, EMBED);
  const int n2 = EMBED*HOPH;
  k_transpose_bf16<<<(n2+255)/256, 256, 0, stream>>>(xi, xiTf, EMBED, HOPH);
  k_copy_bf16<<<(n2+255)/256, 256, 0, stream>>>(xi, xiB, n2);
  const int n3 = 1536*EMBED;
  k_pack_wqk<<<(n3+255)/256, 256, 0, stream>>>(wq, wk, WqkF, WqkB);
  const int n4 = MPAT*EMBED;
  k_patchify<<<(n4+255)/256, 256, 0, stream>>>(x, Pm);

  // ---- patch embedding -> h rows t>=1, then cls + positional ----
  launch_gemm(Pm, patchWt, MPAT, EMBED, EMBED, h, nullptr, patch_b, /*mode=*/2, 0, stream);
  const int n5 = MTOK*EMBED;
  k_init_h<<<(n5+255)/256, 256, 0, stream>>>(h, cls_tok, pos_emb);

  // ---- 6 gradient-descent steps on the energy ----
  for (int t = 0; t < TSTEPS; ++t) {
    k_layernorm<<<(MTOK+7)/8, 256, 0, stream>>>(h, ln_gamma, ln_beta, g, MTOK);
    // Q,K projections (bf16 out)
    launch_gemm(g, WqkF, MTOK, 1536, EMBED, nullptr, QK, nullptr, 0, 0, stream);
    // Hopfield hidden = relu(g @ xi)  (bf16 out)
    launch_gemm(g, xiTf, MTOK, HOPH, EMBED, nullptr, Hhid, nullptr, 0, 1, stream);
    // attention gradient: G = [P K | P^T Q] per head
    k_attention<<<BATCH*NHEADS, 256, ATT_LDS_BYTES, stream>>>(QK, G);
    // h += G @ [wq|wk]^T   and   h += relu(g@xi) @ xi^T
    launch_gemm(G,    WqkB, MTOK, EMBED, 1536, h, nullptr, nullptr, 1, 0, stream);
    launch_gemm(Hhid, xiB,  MTOK, EMBED, HOPH, h, nullptr, nullptr, 1, 0, stream);
  }

  // ---- output head: LN + projection, skip CLS row ----
  k_layernorm<<<(MTOK+7)/8, 256, 0, stream>>>(h, out_gamma, out_beta, g, MTOK);
  launch_gemm(g, outWt, MTOK, EMBED, EMBED, out, nullptr, out_b, /*mode=*/3, 0, stream);
}
